// Head_88261577932902
// MI455X (gfx1250) — compile-verified, matrix operand order checked by reading, not run on hardware
//
#include <hip/hip_runtime.h>

// ---------------- problem constants ----------------
static constexpr int BATCH = 4;
static constexpr int SEQ   = 4096;
static constexpr int EMBED = 1024;
static constexpr int HEAD  = 128;

typedef __attribute__((ext_vector_type(16))) __bf16 v16bf;
typedef __attribute__((ext_vector_type(8)))  float  v8f;
typedef unsigned int   u32;
typedef unsigned short u16;

union Frag {          // one WMMA 16x32 bf16 operand (8 VGPRs per lane)
    v16bf v;
    u32   u[8];
};

// round-to-nearest-even f32 -> bf16 bits
__device__ __forceinline__ u16 f2bf(float f) {
    u32 u = __builtin_bit_cast(u32, f);
    u32 r = u + 0x7FFFu + ((u >> 16) & 1u);
    return (u16)(r >> 16);
}

// K-offset of VGPR j for A/B bf16 fragments (ISA 7.12.2):
// VGPR 0-3 : K = 2j,2j+1   (+8 for lanes 16-31)
// VGPR 4-7 : K = 16+2(j-4) (+8 for lanes 16-31)
__device__ __forceinline__ int koff(int j, int half) {
    return 2 * j + ((j >= 4) ? 8 : 0) + half * 8;
}

__device__ __forceinline__ v8f vzero() {
    v8f z;
#pragma unroll
    for (int i = 0; i < 8; ++i) z[i] = 0.0f;
    return z;
}

// ---------------- kernel 1: x fp32 -> bf16 ----------------
__global__ void convert_x_kernel(const float* __restrict__ x,
                                 u16* __restrict__ xb, int n) {
    int idx = blockIdx.x * blockDim.x + threadIdx.x;
    if (idx < n) xb[idx] = f2bf(x[idx]);
}

// ---------------- kernel 2: W fp32 [E][H] -> bf16 transposed [3][H][E] ----------------
__global__ void convert_w_kernel(const float* __restrict__ Wq,
                                 const float* __restrict__ Wk,
                                 const float* __restrict__ Wv,
                                 u16* __restrict__ wt) {
    int idx = blockIdx.x * blockDim.x + threadIdx.x;
    if (idx >= 3 * HEAD * EMBED) return;
    int m   = idx / (HEAD * EMBED);
    int rem = idx - m * (HEAD * EMBED);
    int h   = rem / EMBED;
    int e   = rem - h * EMBED;
    const float* W = (m == 0) ? Wq : ((m == 1) ? Wk : Wv);
    wt[idx] = f2bf(W[e * HEAD + h]);   // wt[m][h][e]
}

// ---------------- kernel 3: QKV projection (16x128 strip per wave) ----------------
// q,k stored row-major bf16 [B*S][H]; v stored transposed bf16 [B][H][S]
__global__ void __launch_bounds__(32)
qkv_gemm_kernel(const u16* __restrict__ xb,   // [B*S][E] bf16
                const u16* __restrict__ wt,   // [3][H][E] bf16
                u16* __restrict__ qb,
                u16* __restrict__ kb,
                u16* __restrict__ vtb) {
    const int lane = threadIdx.x;
    const int half = lane >> 4;
    const int n    = lane & 15;
    const long srow0 = (long)blockIdx.x * 16;   // global row tile over B*S
    const int  mat   = blockIdx.y;              // 0=q 1=k 2=v
    const u16* wm = wt + (size_t)mat * HEAD * EMBED;

    v8f acc[8];
#pragma unroll
    for (int t = 0; t < 8; ++t) acc[t] = vzero();

    const u16* arow = xb + (size_t)(srow0 + n) * EMBED;   // A row M = lane&15
    const u16* wrow = wm + (size_t)n * EMBED;             // B col N = lane&15

    for (int kc = 0; kc < EMBED; kc += 32) {
        // unconditional speculative prefetch of the next A chunk
        __builtin_prefetch(arow + kc + 32, 0, 1);

        Frag a;
#pragma unroll
        for (int j = 0; j < 8; ++j)
            a.u[j] = *(const u32*)(arow + kc + koff(j, half));

        // double-buffered streaming of B fragments: load t+1 before using t
        Frag bfr[2];
#pragma unroll
        for (int j = 0; j < 8; ++j)
            bfr[0].u[j] = *(const u32*)(wrow + kc + koff(j, half));
#pragma unroll
        for (int t = 0; t < 8; ++t) {
            if (t < 7) {
                const u16* brow = wrow + (size_t)(t + 1) * 16 * EMBED + kc;
#pragma unroll
                for (int j = 0; j < 8; ++j)
                    bfr[(t + 1) & 1].u[j] = *(const u32*)(brow + koff(j, half));
            }
            acc[t] = __builtin_amdgcn_wmma_f32_16x16x32_bf16(
                false, a.v, false, bfr[t & 1].v, (short)0, acc[t], false, false);
        }
    }

    // ---- epilogue: branch on mat ONCE, not per element ----
    if (mat == 2) {
        // v transposed [b][h][s]: lane holds 8 consecutive s -> one b128 store per tile
        const long b  = srow0 >> 12;                 // / SEQ
        const long s0 = (srow0 & (SEQ - 1)) + 8 * half;
#pragma unroll
        for (int t = 0; t < 8; ++t) {
            u32 packed[4];
#pragma unroll
            for (int p = 0; p < 4; ++p) {
                const u32 lo = f2bf(acc[t][2 * p]);
                const u32 hi = f2bf(acc[t][2 * p + 1]);
                packed[p] = lo | (hi << 16);
            }
            u16* dst = vtb + ((size_t)b * HEAD + t * 16 + n) * SEQ + s0;
            *(uint4*)dst = make_uint4(packed[0], packed[1], packed[2], packed[3]);
        }
    } else {
        u16* outp = (mat == 0) ? qb : kb;            // row-major [B*S][H]
#pragma unroll
        for (int t = 0; t < 8; ++t)
#pragma unroll
            for (int r = 0; r < 8; ++r)
                outp[(srow0 + r + 8 * half) * HEAD + t * 16 + n] = f2bf(acc[t][r]);
    }
}

// ---------------- kernel 4: causal flash attention (16 queries per wave) ----------------
__global__ void __launch_bounds__(32)
flash_attn_kernel(const u16* __restrict__ qb,    // [B*S][H]
                  const u16* __restrict__ kb,    // [B*S][H]
                  const u16* __restrict__ vtb,   // [B][H][S]
                  float* __restrict__ out) {     // [B*S][H] fp32
    __shared__ float sS[16][32];                       // score tile (f32)
    __shared__ __align__(16) u16 sP[16][32];           // softmaxed P tile (bf16)
    __shared__ float sAlpha[16];
    __shared__ float sInv[16];

    const int lane = threadIdx.x;
    const int half = lane >> 4;
    const int n    = lane & 15;
    const int b    = blockIdx.y;
    const int q0   = blockIdx.x * 16;
    const float scale = 0.0883883476483184f;           // 1/sqrt(128)

    // load Q A-fragments once (H=128 -> 4 K-chunks of 32)
    Frag qf[4];
    const u16* qrow = qb + ((size_t)b * SEQ + q0 + n) * HEAD;
#pragma unroll
    for (int c = 0; c < 4; ++c)
#pragma unroll
        for (int j = 0; j < 8; ++j)
            qf[c].u[j] = *(const u32*)(qrow + c * 32 + koff(j, half));

    v8f o[8];
#pragma unroll
    for (int t = 0; t < 8; ++t) o[t] = vzero();
    float mrow = -1e30f, lrow = 0.0f;                  // valid on lanes 0..15

    const int ktMax = (q0 + 15) >> 5;                  // inclusive 32-key tiles
    for (int kt = 0; kt <= ktMax; ++kt) {
        const int kbase = kt * 32;

        // unconditional speculative prefetch of next tile's K rows / V columns
        __builtin_prefetch(kb + ((size_t)b * SEQ + kbase + 32 + n) * HEAD, 0, 1);
        __builtin_prefetch(vtb + ((size_t)b * HEAD + lane * 4) * SEQ + kbase + 32, 0, 1);

        // ---- S = Q * K^T (two 16x16 tiles covering 32 keys) ----
        // double-buffer the K fragments across the 4 H-chunks
        const u16* krow = kb + ((size_t)b * SEQ + kbase + n) * HEAD;
        v8f s0 = vzero(), s1 = vzero();
        Frag k0f[2], k1f[2];
#pragma unroll
        for (int j = 0; j < 8; ++j) {
            k0f[0].u[j] = *(const u32*)(krow + koff(j, half));
            k1f[0].u[j] = *(const u32*)(krow + 16 * HEAD + koff(j, half));
        }
#pragma unroll
        for (int c = 0; c < 4; ++c) {
            if (c < 3) {
                const u16* krn = krow + (c + 1) * 32;
#pragma unroll
                for (int j = 0; j < 8; ++j) {
                    k0f[(c + 1) & 1].u[j] = *(const u32*)(krn + koff(j, half));
                    k1f[(c + 1) & 1].u[j] = *(const u32*)(krn + 16 * HEAD + koff(j, half));
                }
            }
            s0 = __builtin_amdgcn_wmma_f32_16x16x32_bf16(
                false, qf[c].v, false, k0f[c & 1].v, (short)0, s0, false, false);
            s1 = __builtin_amdgcn_wmma_f32_16x16x32_bf16(
                false, qf[c].v, false, k1f[c & 1].v, (short)0, s1, false, false);
        }
#pragma unroll
        for (int r = 0; r < 8; ++r) {
            sS[r + 8 * half][n]      = s0[r] * scale;
            sS[r + 8 * half][n + 16] = s1[r] * scale;
        }
        __syncthreads();

        // ---- online softmax: lane L<16 owns query row L ----
        if (lane < 16) {
            const int row = lane, q = q0 + row;
            float mNew = mrow;
            for (int c2 = 0; c2 < 32; ++c2) {
                float s = (kbase + c2 <= q) ? sS[row][c2] : -1e30f;
                mNew = fmaxf(mNew, s);
            }
            const float alpha = __expf(mrow - mNew);
            float sum = 0.0f;
            for (int c2 = 0; c2 < 32; ++c2) {
                float p = (kbase + c2 <= q) ? __expf(sS[row][c2] - mNew) : 0.0f;
                sum += p;
                sP[row][c2] = f2bf(p);
            }
            lrow = lrow * alpha + sum;
            mrow = mNew;
            sAlpha[row] = alpha;
        }
        __syncthreads();

        // ---- rescale O, then O += P * V ----
        float av[8];
#pragma unroll
        for (int r = 0; r < 8; ++r) av[r] = sAlpha[r + 8 * half];
#pragma unroll
        for (int t = 0; t < 8; ++t)
#pragma unroll
            for (int r = 0; r < 8; ++r) o[t][r] *= av[r];

        Frag pf;                                        // P as A-fragment (16x32)
#pragma unroll
        for (int j = 0; j < 8; ++j)
            pf.u[j] = *(const u32*)&sP[n][koff(j, half)];

        // double-buffered streaming of V fragments
        const u16* vbase = vtb + ((size_t)b * HEAD + n) * SEQ + kbase;
        Frag vf[2];
#pragma unroll
        for (int j = 0; j < 8; ++j)
            vf[0].u[j] = *(const u32*)(vbase + koff(j, half));
#pragma unroll
        for (int t = 0; t < 8; ++t) {
            if (t < 7) {
                const u16* vrn = vbase + (size_t)(t + 1) * 16 * SEQ;
#pragma unroll
                for (int j = 0; j < 8; ++j)
                    vf[(t + 1) & 1].u[j] = *(const u32*)(vrn + koff(j, half));
            }
            o[t] = __builtin_amdgcn_wmma_f32_16x16x32_bf16(
                false, pf.v, false, vf[t & 1].v, (short)0, o[t], false, false);
        }
        __syncthreads();
    }

    // ---- normalize and store ----
    if (lane < 16) sInv[lane] = 1.0f / lrow;
    __syncthreads();
    float inv[8];
#pragma unroll
    for (int r = 0; r < 8; ++r) inv[r] = sInv[r + 8 * half];
#pragma unroll
    for (int t = 0; t < 8; ++t)
#pragma unroll
        for (int r = 0; r < 8; ++r)
            out[((size_t)b * SEQ + q0 + r + 8 * half) * HEAD + t * 16 + n] =
                o[t][r] * inv[r];
}

// ---------------- launcher ----------------
extern "C" void kernel_launch(void* const* d_in, const int* in_sizes, int n_in,
                              void* d_out, int out_size, void* d_ws, size_t ws_size,
                              hipStream_t stream) {
    (void)in_sizes; (void)n_in; (void)out_size; (void)ws_size;
    const float* x  = (const float*)d_in[0];
    const float* Wq = (const float*)d_in[1];
    const float* Wk = (const float*)d_in[2];
    const float* Wv = (const float*)d_in[3];
    float* out = (float*)d_out;

    char*  ws  = (char*)d_ws;
    size_t off = 0;
    auto alloc = [&](size_t bytes) -> void* {
        void* p = ws + off;
        off += (bytes + 255) & ~(size_t)255;
        return p;
    };
    u16* xb  = (u16*)alloc((size_t)BATCH * SEQ * EMBED * 2);   // 32 MB
    u16* wt  = (u16*)alloc((size_t)3 * HEAD * EMBED * 2);      // 768 KB
    u16* qb  = (u16*)alloc((size_t)BATCH * SEQ * HEAD * 2);    // 4 MB
    u16* kb  = (u16*)alloc((size_t)BATCH * SEQ * HEAD * 2);    // 4 MB
    u16* vtb = (u16*)alloc((size_t)BATCH * SEQ * HEAD * 2);    // 4 MB

    {
        const int n = BATCH * SEQ * EMBED;
        convert_x_kernel<<<(n + 255) / 256, 256, 0, stream>>>(x, xb, n);
    }
    {
        const int n = 3 * HEAD * EMBED;
        convert_w_kernel<<<(n + 255) / 256, 256, 0, stream>>>(Wq, Wk, Wv, wt);
    }
    {
        dim3 grid(BATCH * SEQ / 16, 3);
        qkv_gemm_kernel<<<grid, 32, 0, stream>>>(xb, wt, qb, kb, vtb);
    }
    {
        dim3 grid(SEQ / 16, BATCH);
        flash_attn_kernel<<<grid, 32, 0, stream>>>(qb, kb, vtb, out);
    }
}
